// BRC_13176959664607
// MI455X (gfx1250) — matrix operand order repaired and checked by reading, not compiled, not run on hardware
//
#include <hip/hip_runtime.h>

// Problem dims (match reference)
#define I_DIM 512      // input size  (GEMM K)
#define H_DIM 1024     // hidden size (GEMM N)
#define B_DIM 64       // batch
#define T_DIM 512      // sequence length
#define CHUNK 32       // timesteps per chunk (keeps 24MB intermediate in L2)
#define M_CHUNK (CHUNK * B_DIM)   // 2048 GEMM rows per chunk

typedef __attribute__((ext_vector_type(4)))  float  v4f;
typedef __attribute__((ext_vector_type(8)))  float  v8f;
typedef __attribute__((ext_vector_type(4)))  __bf16 v4bf;
typedef __attribute__((ext_vector_type(16))) __bf16 v16bf;

// LDS row stride in bf16 elems: 40 bf16 = 20 words; 20*row mod 64 enumerates
// every multiple of 4 -> bank-friendly b128 fragment reads, 16B-alignable.
#define LDSB 40
#define TILE_ELEMS (128 * LDSB)
#define TILE_BYTES (TILE_ELEMS * 2)       // 10240 B per tile
#define BUF_BYTES  (4 * TILE_BYTES)       // Ah, Al, Bh, Bl

#define KSTAGES (I_DIM / 32)              // 16

union Frag16 { v16bf v; uint4 q[2]; };

// Async memory->LDS copy, GVS form: per-lane 16B, no data VGPRs (ASYNCcnt).
__device__ __forceinline__ void async_b128(unsigned lds_byte_addr,
                                           unsigned voff_bytes,
                                           const void* sbase) {
    asm volatile("global_load_async_to_lds_b128 %0, %1, %2"
                 :: "v"(lds_byte_addr), "v"(voff_bytes), "s"(sbase)
                 : "memory");
}
__device__ __forceinline__ void async_wait8() {
    asm volatile("s_wait_asynccnt 0x8" ::: "memory");
}
__device__ __forceinline__ void async_wait0() {
    asm volatile("s_wait_asynccnt 0x0" ::: "memory");
}

// ---------------------------------------------------------------------------
// f32 -> (bf16 hi, bf16 lo) split:  hi = RNE(x), lo = RNE(x - hi).
// hi*hi + hi*lo + lo*hi recovers ~fp32 product accuracy on the bf16 pipe.
// ---------------------------------------------------------------------------
__global__ __launch_bounds__(256) void cvt_hilo(const float* __restrict__ src,
                                                __bf16* __restrict__ hi,
                                                __bf16* __restrict__ lo, int n4) {
    const int i = blockIdx.x * 256 + threadIdx.x;
    if (i >= n4) return;
    const v4f v = *(const v4f*)(src + (size_t)i * 4);
    v4bf h, l;
    #pragma unroll
    for (int j = 0; j < 4; ++j) {
        const __bf16 hh = (__bf16)v[j];
        h[j] = hh;
        l[j] = (__bf16)(v[j] - (float)hh);
    }
    *(v4bf*)(hi + (size_t)i * 4) = h;
    *(v4bf*)(lo + (size_t)i * 4) = l;
}

// ---------------------------------------------------------------------------
// Fused projection GEMM, bf16x3 split on V_WMMA_F32_16X16X32_BF16, with
// double-buffered GLOBAL_LOAD_ASYNC_TO_LDS_B128 staging (ASYNCcnt pipeline).
// out[g][m][n] = sum_k X[m][k]*U_g[n][k] + b_g[n]
// grid = (M_CHUNK/128, H_DIM/128, 3 gates); block = 256 threads (8 waves).
// Each wave owns a 32(M) x 64(N) strip = 2x4 tiles of 16x16 f32 accumulators.
// ---------------------------------------------------------------------------
__global__ __launch_bounds__(256) void gated_proj_gemm(
    const __bf16* __restrict__ Xh, const __bf16* __restrict__ Xl,
    const __bf16* __restrict__ UhA, const __bf16* __restrict__ UlA,
    const float* __restrict__ bc, const float* __restrict__ ba,
    const float* __restrict__ bhb,
    float* __restrict__ pc, float* __restrict__ pa, float* __restrict__ ph)
{
    const __bf16* Uhg = UhA + (size_t)blockIdx.z * H_DIM * I_DIM;
    const __bf16* Ulg = UlA + (size_t)blockIdx.z * H_DIM * I_DIM;
    const float* bias = (blockIdx.z == 0) ? bc : ((blockIdx.z == 1) ? ba : bhb);
    float* out        = (blockIdx.z == 0) ? pc : ((blockIdx.z == 1) ? pa : ph);

    // [buffer][tile: Ah,Al,Bh,Bl][128 rows x 32 k (+pad)] = 80KB
    __shared__ __bf16 lds[2][4][TILE_ELEMS];

    const int tid  = threadIdx.x;
    const int lane = tid & 31;
    const int wave = tid >> 5;
    const int wm   = wave & 3;          // M quadrant (32 rows)
    const int wn   = wave >> 2;         // N half (64 cols)
    const int m0   = blockIdx.x * 128;
    const int n0   = blockIdx.y * 128;
    const int half = lane >> 4;
    const int l15  = lane & 15;

    // Per-thread staging segments: s in {tid, tid+256}; 16B (8 bf16) each.
    unsigned voffA[2], voffB[2], ldso[2];
    #pragma unroll
    for (int p = 0; p < 2; ++p) {
        const int s = tid + p * 256;
        const int row = s >> 2;
        const int c8  = (s & 3) * 8;
        voffA[p] = (unsigned)(((m0 + row) * I_DIM + c8) * 2);   // bytes into X planes
        voffB[p] = (unsigned)(((n0 + row) * I_DIM + c8) * 2);   // bytes into U planes
        ldso[p]  = (unsigned)((row * LDSB + c8) * 2);           // bytes within a tile
    }
    // Flat->LDS aperture rule: low 32 bits of a generic shared pointer are
    // the wave-relative LDS byte address.
    const unsigned lbase = (unsigned)(unsigned long long)&lds[0][0][0];

    auto issue = [&](int buf, unsigned koff) {
        #pragma unroll
        for (int p = 0; p < 2; ++p) {
            const unsigned lb = lbase + (unsigned)buf * BUF_BYTES + ldso[p];
            async_b128(lb + 0 * TILE_BYTES, voffA[p] + koff, Xh);
            async_b128(lb + 1 * TILE_BYTES, voffA[p] + koff, Xl);
            async_b128(lb + 2 * TILE_BYTES, voffB[p] + koff, Uhg);
            async_b128(lb + 3 * TILE_BYTES, voffB[p] + koff, Ulg);
        }
    };

    const v8f vz = {};
    v8f acc[2][4];
    #pragma unroll
    for (int mi = 0; mi < 2; ++mi)
        #pragma unroll
        for (int ni = 0; ni < 4; ++ni) acc[mi][ni] = vz;

    issue(0, 0);   // prologue: stage 0 into buffer 0

    for (int it = 0; it < KSTAGES; ++it) {
        const int cur = it & 1;
        if (it + 1 < KSTAGES) {
            // Safe: every wave passed the previous trailing barrier before
            // any wave issues into buffer cur^1.
            issue(cur ^ 1, (unsigned)((it + 1) * 64));
            async_wait8();    // stage `it`'s 8 loads done (in-order per wave)
        } else {
            async_wait0();
        }
        __syncthreads();      // all waves' stage-`it` tiles visible

        const __bf16* pAh = lds[cur][0];
        const __bf16* pAl = lds[cur][1];
        const __bf16* pBh = lds[cur][2];
        const __bf16* pBl = lds[cur][3];

        // A 16x32 bf16 layout: lane<16 -> K 0..7 & 16..23; lane>=16 -> 8..15 & 24..31
        Frag16 fah[2], fal[2], fbh[4], fbl[4];
        #pragma unroll
        for (int mi = 0; mi < 2; ++mi) {
            const int r = (wm * 32 + mi * 16 + l15) * LDSB + half * 8;
            fah[mi].q[0] = *(const uint4*)&pAh[r];
            fah[mi].q[1] = *(const uint4*)&pAh[r + 16];
            fal[mi].q[0] = *(const uint4*)&pAl[r];
            fal[mi].q[1] = *(const uint4*)&pAl[r + 16];
        }
        // B 32x16 bf16 layout: lane<16 -> K 0..7 & 16..23; lane>=16 -> 8..15 & 24..31
        #pragma unroll
        for (int ni = 0; ni < 4; ++ni) {
            const int r = (wn * 64 + ni * 16 + l15) * LDSB + half * 8;
            fbh[ni].q[0] = *(const uint4*)&pBh[r];
            fbh[ni].q[1] = *(const uint4*)&pBh[r + 16];
            fbl[ni].q[0] = *(const uint4*)&pBl[r];
            fbl[ni].q[1] = *(const uint4*)&pBl[r + 16];
        }

        // 3-pass split product: lo*hi + hi*lo + hi*hi (f32 accumulate)
        #pragma unroll
        for (int mi = 0; mi < 2; ++mi) {
            #pragma unroll
            for (int ni = 0; ni < 4; ++ni) {
                acc[mi][ni] = __builtin_amdgcn_wmma_f32_16x16x32_bf16(
                    false, fal[mi].v, false, fbh[ni].v, (short)0, acc[mi][ni], false, false);
                acc[mi][ni] = __builtin_amdgcn_wmma_f32_16x16x32_bf16(
                    false, fah[mi].v, false, fbl[ni].v, (short)0, acc[mi][ni], false, false);
                acc[mi][ni] = __builtin_amdgcn_wmma_f32_16x16x32_bf16(
                    false, fah[mi].v, false, fbh[ni].v, (short)0, acc[mi][ni], false, false);
            }
        }
        __syncthreads();      // all waves done reading buffer `cur`
    }

    // Epilogue: bias add + store. C/D layout: (m,n) -> vgpr m%8, lane (m/8)*16+n.
    #pragma unroll
    for (int mi = 0; mi < 2; ++mi) {
        #pragma unroll
        for (int ni = 0; ni < 4; ++ni) {
            const int n   = n0 + wn * 64 + ni * 16 + l15;
            const float bv = bias[n];
            #pragma unroll
            for (int v = 0; v < 8; ++v) {
                const int m = m0 + wm * 32 + mi * 16 + half * 8 + v;
                out[(size_t)m * H_DIM + n] = acc[mi][ni][v] + bv;
            }
        }
    }
}

// ---------------------------------------------------------------------------
// Gated recurrence over one chunk: elementwise in (b,h), each thread scans
// its position privately across CHUNK timesteps; projections are L2-hot.
// ---------------------------------------------------------------------------
__global__ __launch_bounds__(256) void gated_scan(
    const float* __restrict__ pc, const float* __restrict__ pa,
    const float* __restrict__ ph,
    const float* __restrict__ wc, const float* __restrict__ wa,
    float* __restrict__ h_buf, float* __restrict__ y, int t0)
{
    const int idx = blockIdx.x * blockDim.x + threadIdx.x;   // 0 .. B*H-1
    const int hi  = idx & (H_DIM - 1);

    float h = h_buf[idx];
    const float wcv = wc[hi];
    const float wav = wa[hi];

    #pragma unroll 4
    for (int t = 0; t < CHUNK; ++t) {
        const size_t off = (size_t)t * (B_DIM * H_DIM) + idx;
        const float c = 1.0f / (1.0f + __expf(-(pc[off] + wcv * h)));
        const float a = 1.0f + tanhf(pa[off] + wav * h);
        h = c * h + (1.0f - c) * tanhf(ph[off] + a * h);
        y[(size_t)(t0 + t) * (B_DIM * H_DIM) + idx] = h;
    }
    h_buf[idx] = h;
}

extern "C" void kernel_launch(void* const* d_in, const int* in_sizes, int n_in,
                              void* d_out, int out_size, void* d_ws, size_t ws_size,
                              hipStream_t stream) {
    (void)in_sizes; (void)n_in; (void)out_size; (void)ws_size;

    const float* x_seq = (const float*)d_in[0];   // [T, B, I]
    const float* h0    = (const float*)d_in[1];   // [B, H]
    const float* Uc    = (const float*)d_in[2];   // [H, I]
    const float* wc    = (const float*)d_in[3];   // [H]
    const float* bc    = (const float*)d_in[4];   // [H]
    const float* Ua    = (const float*)d_in[5];
    const float* wa    = (const float*)d_in[6];
    const float* ba    = (const float*)d_in[7];
    const float* Uh    = (const float*)d_in[8];
    const float* bh    = (const float*)d_in[9];

    float* y          = (float*)d_out;                          // [T, B, H]
    float* h_last_out = y + (size_t)T_DIM * B_DIM * H_DIM;      // [B, H]

    const size_t proj_elems = (size_t)CHUNK * B_DIM * H_DIM;    // 2M floats
    const size_t u_elems    = (size_t)H_DIM * I_DIM;            // 512K
    const size_t x_elems    = (size_t)M_CHUNK * I_DIM;          // 1M

    float* pc    = (float*)d_ws;
    float* pa    = pc + proj_elems;
    float* ph    = pa + proj_elems;
    float* h_buf = ph + proj_elems;                              // [B*H]
    __bf16* xhi  = (__bf16*)(h_buf + (size_t)B_DIM * H_DIM);
    __bf16* xlo  = xhi + x_elems;
    __bf16* uhi  = xlo + x_elems;                                // 3 gates contiguous
    __bf16* ulo  = uhi + 3 * u_elems;

    hipMemcpyAsync(h_buf, h0, (size_t)B_DIM * H_DIM * sizeof(float),
                   hipMemcpyDeviceToDevice, stream);

    // U hi/lo split once per launch (reused by every chunk)
    const int un4 = (int)(u_elems / 4);
    cvt_hilo<<<dim3(un4 / 256), 256, 0, stream>>>(Uc, uhi,               ulo,               un4);
    cvt_hilo<<<dim3(un4 / 256), 256, 0, stream>>>(Ua, uhi + u_elems,     ulo + u_elems,     un4);
    cvt_hilo<<<dim3(un4 / 256), 256, 0, stream>>>(Uh, uhi + 2 * u_elems, ulo + 2 * u_elems, un4);

    const dim3 ggrid(M_CHUNK / 128, H_DIM / 128, 3);
    const dim3 sgrid((B_DIM * H_DIM) / 256);
    const int xn4 = (int)(x_elems / 4);

    for (int chunk = 0; chunk < T_DIM / CHUNK; ++chunk) {
        const float* X = x_seq + (size_t)chunk * CHUNK * B_DIM * I_DIM;
        cvt_hilo<<<dim3(xn4 / 256), 256, 0, stream>>>(X, xhi, xlo, xn4);
        gated_proj_gemm<<<ggrid, 256, 0, stream>>>(xhi, xlo, uhi, ulo,
                                                   bc, ba, bh, pc, pa, ph);
        gated_scan<<<sgrid, 256, 0, stream>>>(pc, pa, ph, wc, wa, h_buf, y,
                                              chunk * CHUNK);
    }

    hipMemcpyAsync(h_last_out, h_buf, (size_t)B_DIM * H_DIM * sizeof(float),
                   hipMemcpyDeviceToDevice, stream);
}
